// UserSeqMergeEncoder_32109175505433
// MI455X (gfx1250) — matrix operand order repaired and checked by reading, not compiled, not run on hardware
//
#include <hip/hip_runtime.h>
#include <hip/hip_bf16.h>

// ---------------------------------------------------------------------------
// UserSeqMergeEncoder for MI455X (gfx1250): bf16 WMMA everywhere, ragged
// flash-attention with analytic zero-attention slot, fused epilogues.
// Round 2: 32x64 register-tiled WMMA GEMM (1.5 loads/wmma), Q pre-scaled by
// 1/sqrt(HD), global prefetch in attention inner loop.
// ---------------------------------------------------------------------------

typedef unsigned short u16;
typedef unsigned int   u32;
typedef __attribute__((ext_vector_type(16))) __bf16 v16bf;
typedef __attribute__((ext_vector_type(8)))  float  v8f;
typedef __attribute__((ext_vector_type(4)))  u32    v4u;
typedef __attribute__((ext_vector_type(2)))  u32    v2u;
typedef __attribute__((ext_vector_type(4)))  float  v4f;

#define DHID   256
#define NHEAD  4
#define HD     64
#define NBATCH 256

__device__ __forceinline__ u16 f2bf(float x) {
  u32 u = __builtin_bit_cast(u32, x);
  u32 r = u + 0x7FFFu + ((u >> 16) & 1u);   // round-to-nearest-even
  return (u16)(r >> 16);
}

__device__ __forceinline__ v2u pack4(v4f x) {
  v2u r;
  r[0] = (u32)f2bf(x[0]) | ((u32)f2bf(x[1]) << 16);
  r[1] = (u32)f2bf(x[2]) | ((u32)f2bf(x[3]) << 16);
  return r;
}

union FragU { v4u q[2]; v16bf bf; };

// Build a 16xbf16 WMMA fragment from two 16-byte chunks.
__device__ __forceinline__ v16bf mk_frag(const u16* p0, const u16* p1) {
  FragU u;
  u.q[0] = *(const v4u*)p0;
  u.q[1] = *(const v4u*)p1;
  return u.bf;
}

__device__ __forceinline__ float gelu_exact(float x) {
  return 0.5f * x * (1.0f + erff(x * 0.70710678118654752440f));
}

// ---------------------------------------------------------------------------
// 0) scan over seq lengths -> starts[], 16-aligned padded starts spad[]
// ---------------------------------------------------------------------------
__global__ void scan_kernel(const int* __restrict__ lens,
                            int* __restrict__ starts,
                            int* __restrict__ spad) {
  __shared__ int s1[NBATCH];
  __shared__ int s2[NBATCH];
  int t = threadIdx.x;
  int L = lens[t];
  int Lp = (L + 15) & ~15;
  s1[t] = L;
  s2[t] = Lp;
  __syncthreads();
  for (int off = 1; off < NBATCH; off <<= 1) {
    int a = (t >= off) ? s1[t - off] : 0;
    int b = (t >= off) ? s2[t - off] : 0;
    __syncthreads();
    s1[t] += a;
    s2[t] += b;
    __syncthreads();
  }
  starts[t] = s1[t] - L;
  spad[t]   = s2[t] - Lp;
}

// ---------------------------------------------------------------------------
// 0b) token -> (batch, local position) maps
// ---------------------------------------------------------------------------
__global__ void fill_maps_kernel(const int* __restrict__ lens,
                                 const int* __restrict__ starts,
                                 int* __restrict__ batch_of,
                                 int* __restrict__ localpos) {
  int b = blockIdx.x;
  int L = lens[b];
  int st = starts[b];
  for (int i = threadIdx.x; i < L; i += blockDim.x) {
    batch_of[st + i] = b;
    localpos[st + i] = i;
  }
}

// ---------------------------------------------------------------------------
// 1) q/k/v_cat = emb + pos_table  (fp32 q_cat kept for residual; bf16 copies
//    for WMMA GEMMs; q_cat bf16 goes directly to right half of merge input)
// ---------------------------------------------------------------------------
__global__ void prep_kernel(const float* __restrict__ qe,
                            const float* __restrict__ ke,
                            const float* __restrict__ ve,
                            const float* __restrict__ pos,
                            const int* __restrict__ localpos,
                            float* __restrict__ qcatf,
                            u16* __restrict__ mcat,     // [L1, 512]
                            u16* __restrict__ kcat,     // [L1, 256]
                            u16* __restrict__ vcat) {   // [L1, 256]
  int t = blockIdx.x;
  int c = threadIdx.x * 4;
  int lp = localpos[t];
  v4f pq = *(const v4f*)(pos + (size_t)(lp + 1) * DHID + c);
  v4f pk = *(const v4f*)(pos + (size_t)lp * DHID + c);
  v4f q = *(const v4f*)(qe + (size_t)t * DHID + c) + pq;
  v4f k = *(const v4f*)(ke + (size_t)t * DHID + c) + pk;
  v4f v = *(const v4f*)(ve + (size_t)t * DHID + c) + pk;
  *(v4f*)(qcatf + (size_t)t * DHID + c) = q;
  *(v2u*)(mcat + (size_t)t * 512 + 256 + c) = pack4(q);
  *(v2u*)(kcat + (size_t)t * DHID + c) = pack4(k);
  *(v2u*)(vcat + (size_t)t * DHID + c) = pack4(v);
}

// ---------------------------------------------------------------------------
// 2) weight pack: W[K][N] fp32 -> Wt[N][K] bf16 (B-fragments become two b128)
// ---------------------------------------------------------------------------
__global__ void pack_weight_kernel(const float* __restrict__ W,
                                   u16* __restrict__ Wt, int Kd, int Nd) {
  int idx = blockIdx.x * 256 + threadIdx.x;
  if (idx >= Kd * Nd) return;
  int k = idx / Nd, n = idx - k * Nd;
  Wt[(size_t)n * Kd + k] = f2bf(W[idx]);
}

// ---------------------------------------------------------------------------
// generic zero fill
// ---------------------------------------------------------------------------
__global__ void fill_zero_kernel(u32* __restrict__ p, long n) {
  long i = (long)blockIdx.x * blockDim.x + threadIdx.x;
  long stride = (long)gridDim.x * blockDim.x;
  for (; i < n; i += stride) p[i] = 0u;
}

// ---------------------------------------------------------------------------
// 3) WMMA GEMM: out[M,256] = A[M,Kd](bf16, stride lda) @ Wt^T + bias, *oscale
//    Register tile per wave: 32 rows x 64 cols = 8 wmma per k-step from
//    12 b128 loads (2 A-frags reused over 4 B-frags, B reused over 2 M).
//    block = 128 threads = 4 waves -> 128 x 64 block tile.
//    mode 0: store bf16   mode 1: store fp32   mode 2: fp32 + exact GELU
// ---------------------------------------------------------------------------
__global__ void gemm_kernel(const u16* __restrict__ A, int lda,
                            const u16* __restrict__ Wt,
                            const float* __restrict__ bias,
                            int M, int Kd,
                            u16* __restrict__ outB, float* __restrict__ outF,
                            int mode, float oscale) {
  int l = threadIdx.x & 31;
  int wave = threadIdx.x >> 5;
  int lm = l & 15, half = l >> 4;
  int rm = (blockIdx.x * 4 + wave) * 32;     // 32-row wave tile
  if (rm >= M) return;
  int cn = blockIdx.y * 64;                  // 64-col wave tile

  int r0 = rm + lm;      if (r0 > M - 1) r0 = M - 1;
  int r1 = rm + 16 + lm; if (r1 > M - 1) r1 = M - 1;
  const u16* ap0 = A + (size_t)r0 * lda;
  const u16* ap1 = A + (size_t)r1 * lda;
  const u16* bp0 = Wt + (size_t)(cn + 0 * 16 + lm) * Kd;
  const u16* bp1 = Wt + (size_t)(cn + 1 * 16 + lm) * Kd;
  const u16* bp2 = Wt + (size_t)(cn + 2 * 16 + lm) * Kd;
  const u16* bp3 = Wt + (size_t)(cn + 3 * 16 + lm) * Kd;

  v8f acc[2][4];
#pragma unroll
  for (int mi = 0; mi < 2; ++mi)
#pragma unroll
    for (int j = 0; j < 4; ++j)
#pragma unroll
      for (int i = 0; i < 8; ++i) acc[mi][j][i] = 0.f;

  int nk = Kd >> 5;
#pragma unroll 2
  for (int kt = 0; kt < nk; ++kt) {
    int ca = kt * 32 + half * 16;
    int cb = kt * 32 + half * 16;
    v16bf a0 = mk_frag(ap0 + ca, ap0 + ca + 8);
    v16bf a1 = mk_frag(ap1 + ca, ap1 + ca + 8);
    v16bf b0 = mk_frag(bp0 + cb, bp0 + cb + 8);
    v16bf b1 = mk_frag(bp1 + cb, bp1 + cb + 8);
    v16bf b2 = mk_frag(bp2 + cb, bp2 + cb + 8);
    v16bf b3 = mk_frag(bp3 + cb, bp3 + cb + 8);
    acc[0][0] = __builtin_amdgcn_wmma_f32_16x16x32_bf16(false, a0, false, b0, (short)0, acc[0][0], false, false);
    acc[0][1] = __builtin_amdgcn_wmma_f32_16x16x32_bf16(false, a0, false, b1, (short)0, acc[0][1], false, false);
    acc[0][2] = __builtin_amdgcn_wmma_f32_16x16x32_bf16(false, a0, false, b2, (short)0, acc[0][2], false, false);
    acc[0][3] = __builtin_amdgcn_wmma_f32_16x16x32_bf16(false, a0, false, b3, (short)0, acc[0][3], false, false);
    acc[1][0] = __builtin_amdgcn_wmma_f32_16x16x32_bf16(false, a1, false, b0, (short)0, acc[1][0], false, false);
    acc[1][1] = __builtin_amdgcn_wmma_f32_16x16x32_bf16(false, a1, false, b1, (short)0, acc[1][1], false, false);
    acc[1][2] = __builtin_amdgcn_wmma_f32_16x16x32_bf16(false, a1, false, b2, (short)0, acc[1][2], false, false);
    acc[1][3] = __builtin_amdgcn_wmma_f32_16x16x32_bf16(false, a1, false, b3, (short)0, acc[1][3], false, false);
  }

#pragma unroll
  for (int mi = 0; mi < 2; ++mi) {
#pragma unroll
    for (int j = 0; j < 4; ++j) {
      int col = cn + j * 16 + lm;
      float bb = bias[col];
#pragma unroll
      for (int i = 0; i < 8; ++i) {
        int row = rm + mi * 16 + i + 8 * half;
        if (row >= M) continue;
        float v = (acc[mi][j][i] + bb) * oscale;
        size_t o = (size_t)row * DHID + col;
        if (mode == 0)       outB[o] = f2bf(v);
        else if (mode == 1)  outF[o] = v;
        else                 outF[o] = gelu_exact(v);
      }
    }
  }
}

// ---------------------------------------------------------------------------
// 4) V transpose into [d][padded token col] so ctx B-fragments are contiguous
// ---------------------------------------------------------------------------
__global__ void transpose_v_kernel(const u16* __restrict__ Vb,
                                   const int* __restrict__ batch_of,
                                   const int* __restrict__ localpos,
                                   const int* __restrict__ spad,
                                   u16* __restrict__ Vt, int LP) {
  int t = blockIdx.x;
  int c = threadIdx.x * 4;
  int b = batch_of[t];
  int col = spad[b] + localpos[t];
  v2u d = *(const v2u*)(Vb + (size_t)t * DHID + c);
  Vt[(size_t)(c + 0) * LP + col] = (u16)(d[0] & 0xFFFFu);
  Vt[(size_t)(c + 1) * LP + col] = (u16)(d[0] >> 16);
  Vt[(size_t)(c + 2) * LP + col] = (u16)(d[1] & 0xFFFFu);
  Vt[(size_t)(c + 3) * LP + col] = (u16)(d[1] >> 16);
}

// ---------------------------------------------------------------------------
// 5) ragged flash attention with zero-attention slot (M0=0, L0=1, ctx0=0).
//    grid = (B, H); block = 128 (4 waves); one q-tile per wave iteration.
//    Q is pre-scaled by 1/sqrt(HD) in its projection epilogue.
//    Per 32-key chunk: 4 wmma scores + online softmax + 4 wmma ctx.
// ---------------------------------------------------------------------------
__global__ void attn_kernel(const u16* __restrict__ Qb,
                            const u16* __restrict__ Kb,
                            const u16* __restrict__ Vt, int LP,
                            const int* __restrict__ starts,
                            const int* __restrict__ spad,
                            const int* __restrict__ lens,
                            u16* __restrict__ ctxb, int L1v) {
  __shared__ u16 plds[4][16 * 32];   // per-wave probs staging (16 q x 32 keys)
  int b = blockIdx.x, h = blockIdx.y;
  int l = threadIdx.x & 31, wave = threadIdx.x >> 5;
  int lm = l & 15, half = l >> 4;
  int Lb = lens[b], st = starts[b], sp = spad[b];
  int nqt = (Lb + 15) >> 4;
  int nkb = (Lb + 31) >> 5;
  u16* P = plds[wave];

  for (int qt = wave; qt < nqt; qt += 4) {
    int qrow = st + qt * 16 + lm;
    if (qrow > L1v - 1) qrow = L1v - 1;
    const u16* qp = Qb + (size_t)qrow * DHID + h * HD;
    v16bf a0 = mk_frag(qp + half * 16, qp + half * 16 + 8);            // d 0..31
    v16bf a1 = mk_frag(qp + 32 + half * 16, qp + 32 + half * 16 + 8);  // d 32..63

    v8f acc[4];
    v8f Mv, Lv;
#pragma unroll
    for (int i = 0; i < 8; ++i) {
      acc[0][i] = 0.f; acc[1][i] = 0.f; acc[2][i] = 0.f; acc[3][i] = 0.f;
      Mv[i] = 0.f;   // zero-attention slot: score 0
      Lv[i] = 1.f;   // exp(0 - 0)
    }

    for (int kb = 0; kb < nkb; ++kb) {
      // prefetch next chunk of K rows and V columns while this one computes
      if (kb + 1 < nkb) {
        int pk = st + (kb + 1) * 32 + lm;
        if (pk > L1v - 1) pk = L1v - 1;
        __builtin_prefetch(Kb + (size_t)pk * DHID + h * HD, 0, 3);
        __builtin_prefetch(Vt + (size_t)(h * HD + lm) * LP + sp + (kb + 1) * 32, 0, 3);
      }
      v8f s[2];
#pragma unroll
      for (int t = 0; t < 2; ++t) {
        int key = kb * 32 + t * 16 + lm;
        int krow = st + key;
        if (krow > L1v - 1) krow = L1v - 1;
        const u16* kp = Kb + (size_t)krow * DHID + h * HD;
        v16bf b0 = mk_frag(kp + half * 16, kp + half * 16 + 8);
        v16bf b1 = mk_frag(kp + 32 + half * 16, kp + 32 + half * 16 + 8);
        v8f sc = {0.f, 0.f, 0.f, 0.f, 0.f, 0.f, 0.f, 0.f};
        sc = __builtin_amdgcn_wmma_f32_16x16x32_bf16(false, a0, false, b0,
                                                     (short)0, sc, false, false);
        sc = __builtin_amdgcn_wmma_f32_16x16x32_bf16(false, a1, false, b1,
                                                     (short)0, sc, false, false);
        if (key >= Lb) {
#pragma unroll
          for (int i = 0; i < 8; ++i) sc[i] = -1.0e30f;
        }
        s[t] = sc;
      }
      // --- online softmax: row stats via shuffles over the 16 key lanes ---
      v8f tmax;
#pragma unroll
      for (int i = 0; i < 8; ++i) tmax[i] = fmaxf(s[0][i], s[1][i]);
#pragma unroll
      for (int m = 1; m <= 8; m <<= 1)
#pragma unroll
        for (int i = 0; i < 8; ++i)
          tmax[i] = fmaxf(tmax[i], __shfl_xor(tmax[i], m, 32));
      v8f f;
#pragma unroll
      for (int i = 0; i < 8; ++i) {
        float Mn = fmaxf(Mv[i], tmax[i]);
        f[i] = __expf(Mv[i] - Mn);
        Mv[i] = Mn;
        s[0][i] = __expf(s[0][i] - Mn);
        s[1][i] = __expf(s[1][i] - Mn);
      }
      v8f rs;
#pragma unroll
      for (int i = 0; i < 8; ++i) rs[i] = s[0][i] + s[1][i];
#pragma unroll
      for (int m = 1; m <= 8; m <<= 1)
#pragma unroll
        for (int i = 0; i < 8; ++i) rs[i] += __shfl_xor(rs[i], m, 32);
#pragma unroll
      for (int i = 0; i < 8; ++i) Lv[i] = Lv[i] * f[i] + rs[i];
#pragma unroll
      for (int j = 0; j < 4; ++j)
#pragma unroll
        for (int i = 0; i < 8; ++i) acc[j][i] *= f[i];

      // --- stage probs to LDS, reload in A-fragment order ---
#pragma unroll
      for (int i = 0; i < 8; ++i) {
        int row = i + 8 * half;
        P[row * 32 + lm]      = f2bf(s[0][i]);
        P[row * 32 + 16 + lm] = f2bf(s[1][i]);
      }
      asm volatile("s_wait_dscnt 0" ::: "memory");
      const u16* pr = P + lm * 32;
      v16bf pa = mk_frag(pr + half * 16, pr + half * 16 + 8);

      // --- ctx += probs @ V (V pre-transposed: contiguous along keys) ---
#pragma unroll
      for (int j = 0; j < 4; ++j) {
        const u16* vp = Vt + (size_t)(h * HD + j * 16 + lm) * LP
                        + sp + kb * 32 + half * 16;
        v16bf vb = mk_frag(vp, vp + 8);
        acc[j] = __builtin_amdgcn_wmma_f32_16x16x32_bf16(false, pa, false, vb,
                                                         (short)0, acc[j],
                                                         false, false);
      }
    }
    // --- epilogue: normalize by denominator and store valid rows ---
#pragma unroll
    for (int i = 0; i < 8; ++i) {
      int rq = qt * 16 + i + 8 * half;
      if (rq >= Lb) continue;
      float inv = 1.0f / Lv[i];
      size_t orow = (size_t)(st + rq) * DHID + h * HD;
#pragma unroll
      for (int j = 0; j < 4; ++j)
        ctxb[orow + j * 16 + lm] = f2bf(acc[j][i] * inv);
    }
  }
}

// ---------------------------------------------------------------------------
// 6) residual + LayerNorm; writes bf16 h into left half of merge input.
//    One wave (32 lanes) per row, shuffle reductions only.
// ---------------------------------------------------------------------------
__global__ void ln_kernel(const float* __restrict__ of,
                          const float* __restrict__ qcatf,
                          const float* __restrict__ g,
                          const float* __restrict__ bt,
                          u16* __restrict__ mcat) {
  int r = blockIdx.x, t = threadIdx.x;   // 32 threads
  const float* po = of + (size_t)r * DHID + t * 8;
  const float* pq = qcatf + (size_t)r * DHID + t * 8;
  v4f a0 = *(const v4f*)po + *(const v4f*)pq;
  v4f a1 = *(const v4f*)(po + 4) + *(const v4f*)(pq + 4);
  float s = 0.f, ss = 0.f;
#pragma unroll
  for (int i = 0; i < 4; ++i) {
    s += a0[i] + a1[i];
    ss += a0[i] * a0[i] + a1[i] * a1[i];
  }
#pragma unroll
  for (int m = 1; m < 32; m <<= 1) {
    s += __shfl_xor(s, m, 32);
    ss += __shfl_xor(ss, m, 32);
  }
  float mu = s * (1.0f / 256.0f);
  float var = ss * (1.0f / 256.0f) - mu * mu;
  float inv = rsqrtf(var + 1e-12f);
  v4f g0 = *(const v4f*)(g + t * 8), g1 = *(const v4f*)(g + t * 8 + 4);
  v4f b0 = *(const v4f*)(bt + t * 8), b1 = *(const v4f*)(bt + t * 8 + 4);
  v4f y0, y1;
#pragma unroll
  for (int i = 0; i < 4; ++i) {
    y0[i] = (a0[i] - mu) * inv * g0[i] + b0[i];
    y1[i] = (a1[i] - mu) * inv * g1[i] + b1[i];
  }
  v4u w;
  v2u w0 = pack4(y0), w1 = pack4(y1);
  w[0] = w0[0]; w[1] = w0[1]; w[2] = w1[0]; w[3] = w1[1];
  *(v4u*)(mcat + (size_t)r * 512 + t * 8) = w;
}

// ---------------------------------------------------------------------------
// host launcher
// ---------------------------------------------------------------------------
extern "C" void kernel_launch(void* const* d_in, const int* in_sizes, int n_in,
                              void* d_out, int out_size, void* d_ws, size_t ws_size,
                              hipStream_t stream) {
  const float* qe  = (const float*)d_in[0];
  const float* ke  = (const float*)d_in[1];
  const float* ve  = (const float*)d_in[2];
  const float* pos = (const float*)d_in[3];
  const float* wq  = (const float*)d_in[4];
  const float* bq  = (const float*)d_in[5];
  const float* wk  = (const float*)d_in[6];
  const float* bk  = (const float*)d_in[7];
  const float* wv  = (const float*)d_in[8];
  const float* bv  = (const float*)d_in[9];
  const float* wo  = (const float*)d_in[10];
  const float* bo  = (const float*)d_in[11];
  const float* lng = (const float*)d_in[12];
  const float* lnb = (const float*)d_in[13];
  const float* wm  = (const float*)d_in[14];
  const float* bm  = (const float*)d_in[15];
  const int* lens  = (const int*)d_in[16];

  const int L1 = in_sizes[0] / DHID;                // 32896
  const int LP = ((L1 + 16 * NBATCH + 127) & ~127); // padded Vt column count

  // ---- carve workspace ----
  char* w = (char*)d_ws;
  auto alloc = [&](size_t bytes) {
    char* p = w;
    w += (bytes + 255) & ~(size_t)255;
    return p;
  };
  int*   starts   = (int*)alloc((size_t)NBATCH * 4);
  int*   spad     = (int*)alloc((size_t)NBATCH * 4);
  int*   batch_of = (int*)alloc((size_t)L1 * 4);
  int*   localpos = (int*)alloc((size_t)L1 * 4);
  float* qcatf    = (float*)alloc((size_t)L1 * DHID * 4);
  u16*   kcat     = (u16*)alloc((size_t)L1 * DHID * 2);
  u16*   vcat     = (u16*)alloc((size_t)L1 * DHID * 2);
  u16*   mcat     = (u16*)alloc((size_t)L1 * 512 * 2);  // [h | q_cat] bf16
  u16*   Qb       = (u16*)alloc((size_t)L1 * DHID * 2);
  u16*   Kb       = (u16*)alloc((size_t)L1 * DHID * 2);
  u16*   Vb       = (u16*)alloc((size_t)L1 * DHID * 2);
  u16*   Vt       = (u16*)alloc((size_t)DHID * LP * 2);
  u16*   ctxb     = (u16*)alloc((size_t)L1 * DHID * 2);
  float* of       = (float*)alloc((size_t)L1 * DHID * 4);
  u16*   wtq      = (u16*)alloc((size_t)DHID * DHID * 2);
  u16*   wtk      = (u16*)alloc((size_t)DHID * DHID * 2);
  u16*   wtv      = (u16*)alloc((size_t)DHID * DHID * 2);
  u16*   wto      = (u16*)alloc((size_t)DHID * DHID * 2);
  u16*   wtm      = (u16*)alloc((size_t)512 * DHID * 2);

  // 0) scan + maps
  scan_kernel<<<1, NBATCH, 0, stream>>>(lens, starts, spad);
  fill_maps_kernel<<<NBATCH, 256, 0, stream>>>(lens, starts, batch_of, localpos);

  // 1) position-embedded token tensors
  prep_kernel<<<L1, 64, 0, stream>>>(qe, ke, ve, pos, localpos,
                                     qcatf, mcat, kcat, vcat);

  // 2) weight packs (transpose + bf16)
  pack_weight_kernel<<<(DHID * DHID + 255) / 256, 256, 0, stream>>>(wq, wtq, DHID, DHID);
  pack_weight_kernel<<<(DHID * DHID + 255) / 256, 256, 0, stream>>>(wk, wtk, DHID, DHID);
  pack_weight_kernel<<<(DHID * DHID + 255) / 256, 256, 0, stream>>>(wv, wtv, DHID, DHID);
  pack_weight_kernel<<<(DHID * DHID + 255) / 256, 256, 0, stream>>>(wo, wto, DHID, DHID);
  pack_weight_kernel<<<(512 * DHID + 255) / 256, 256, 0, stream>>>(wm, wtm, 512, DHID);

  // 3) Q/K/V projections (bf16 WMMA GEMMs). Q pre-scaled by 1/sqrt(HD).
  dim3 gg((L1 + 127) / 128, DHID / 64);
  gemm_kernel<<<gg, 128, 0, stream>>>(mcat + 256, 512, wtq, bq, L1, DHID, Qb, nullptr, 0, 0.125f);
  gemm_kernel<<<gg, 128, 0, stream>>>(kcat, DHID, wtk, bk, L1, DHID, Kb, nullptr, 0, 1.0f);
  gemm_kernel<<<gg, 128, 0, stream>>>(vcat, DHID, wtv, bv, L1, DHID, Vb, nullptr, 0, 1.0f);

  // 4) V transpose into padded [d][token] layout (zero pad cols first)
  fill_zero_kernel<<<2048, 256, 0, stream>>>((u32*)Vt, (long)DHID * LP / 2);
  transpose_v_kernel<<<L1, 64, 0, stream>>>(Vb, batch_of, localpos, spad, Vt, LP);

  // 5) ragged flash attention with zero-attention slot
  attn_kernel<<<dim3(NBATCH, NHEAD), 128, 0, stream>>>(Qb, Kb, Vt, LP,
                                                       starts, spad, lens,
                                                       ctxb, L1);

  // 6) output projection (fp32 out) + residual/LayerNorm -> bf16 h
  gemm_kernel<<<gg, 128, 0, stream>>>(ctxb, DHID, wto, bo, L1, DHID, nullptr, of, 1, 1.0f);
  ln_kernel<<<L1, 32, 0, stream>>>(of, qcatf, lng, lnb, mcat);

  // 7) merge GEMM over K=512 with fused exact GELU -> d_out fp32
  gemm_kernel<<<gg, 128, 0, stream>>>(mcat, 512, wtm, bm, L1, 512,
                                      nullptr, (float*)d_out, 2, 1.0f);
}
